// RNNModule_60825326846598
// MI455X (gfx1250) — matrix operand order repaired and checked by reading, not compiled, not run on hardware
//
#include <hip/hip_runtime.h>
#include <cmath>

// Problem constants (match reference).
#define T_STEPS 1024
#define BATCH   64
#define NDIM    512
#define ALPHA   0.1f

// W K-slice split: 9 slices live in LDS (288 KB), 7 slices in VGPRs (112/wave).
#define KT_LDS  9
#define KT_REG  7

typedef __attribute__((ext_vector_type(16))) __bf16 v16bf;
typedef __attribute__((ext_vector_type(8)))  __bf16 v8bf;
typedef __attribute__((ext_vector_type(8)))  float  v8f;

#if __has_builtin(__builtin_amdgcn_tanhf)
  #define FAST_TANH(x) __builtin_amdgcn_tanhf(x)
#elif __has_builtin(__builtin_amdgcn_tanh_f32)
  #define FAST_TANH(x) __builtin_amdgcn_tanh_f32(x)
#else
  #define FAST_TANH(x) tanhf(x)
#endif

// ---------------------------------------------------------------------------
// Pack W_eff = mask ? W_rec : 0 into bf16 WMMA B-fragment layout.
// out[m,n] = sum_k r[m,k] * W[n,k]  =>  B[k,n] = W[n,k].
// B layout used here (contiguous K) (16-bit, 32x16 KxN, wave32): lane L holds column n = nt*16+L%16,
//   elements e=0..15 map to K = kt*32 + (L>=16 ? 16 : 0) + e.
// Memory: Wb[frag = nt*16+kt][lane][e] contiguous -> one 32 B vector per lane.
// ---------------------------------------------------------------------------
__global__ void pack_w_kernel(const float* __restrict__ W,
                              const unsigned char* __restrict__ mask,
                              __bf16* __restrict__ Wb) {
    int id = blockIdx.x * blockDim.x + threadIdx.x;   // 0 .. 512*512-1
    if (id >= NDIM * NDIM) return;
    int f    = id >> 9;          // fragment index = nt*16 + kt
    int rem  = id & 511;
    int lane = rem >> 4;
    int e    = rem & 15;
    int nt = f >> 4;
    int kt = f & 15;
    int n = nt * 16 + (lane & 15);
    int k = kt * 32 + ((lane >> 4) << 4) + e;
    float w = mask[n * NDIM + k] ? W[n * NDIM + k] : 0.0f;
    Wb[id] = (__bf16)w;
}

// ---------------------------------------------------------------------------
// Persistent RNN kernel: 4 workgroups, one per 16-row batch group (the
// recurrence is independent per batch row -> no cross-WG sync, only two
// s_barriers per step). Storage plan per WGP:
//   - x state:      in registers (WMMA C layout, v8f per tile)
//   - W (bf16):     9/16 K-slices in LDS (288 KB), 7/16 in VGPRs (112/wave)
//   - r = tanh(x):  bf16 in LDS (A-fragment friendly row-major)
// ---------------------------------------------------------------------------
__global__ __launch_bounds__(512) void rnn_kernel(
    const float* __restrict__ inputs,   // [T, B, N] fp32
    const float* __restrict__ bias,     // [N] fp32
    const float* __restrict__ init_x,   // [N] fp32
    const __bf16* __restrict__ Wb,      // packed B fragments (512 KB, L2-hot)
    float* __restrict__ out)            // [T+1, B, N] fp32
{
    // 288 KB W + ~16.3 KB r  (<= 320 KB WGP LDS)
    __shared__ __align__(32) __bf16 w_lds[16 * 2 * KT_LDS * 512];
    __shared__ __align__(32) __bf16 r_lds[16][NDIM + 8];

    const int tid  = threadIdx.x;
    const int g    = blockIdx.x;        // batch group: global rows g*16 .. +15
    const int wave = tid >> 5;          // 0..15
    const int lane = tid & 31;

    const int nt0     = wave * 2;       // each wave owns 2 N-tiles (32 cols)
    const int nt1     = wave * 2 + 1;
    const int mloc_lo = (lane < 16) ? 0 : 8;   // C layout: vgpr i -> M = mloc_lo+i
    const int ncol    = lane & 15;             // C layout: N = tile*16 + lane%16
    const int arow    = lane & 15;             // A layout: M = lane%16
    const int akoff   = (lane >> 4) << 4;      // A layout: upper half-wave K+16
    const int n0 = nt0 * 16 + ncol;
    const int n1 = nt1 * 16 + ncol;

    // ---- one-time: stage 9 K-slices of this wave's W tiles into LDS ----
    #pragma unroll
    for (int tau = 0; tau < 2; ++tau) {
        const int nt = (tau == 0) ? nt0 : nt1;
        #pragma unroll
        for (int kt = 0; kt < KT_LDS; ++kt) {
            const unsigned dst = (((unsigned)(wave * 2 + tau) * KT_LDS + kt) * 32
                                  + lane) * 16;
            *(v16bf*)&w_lds[dst] =
                *(const v16bf*)(Wb + ((size_t)(nt * 16 + kt) << 9) + lane * 16);
        }
    }
    // ---- one-time: 7 K-slices of W into registers (112 VGPRs/wave) ----
    v16bf wreg0[KT_REG], wreg1[KT_REG];
    #pragma unroll
    for (int j = 0; j < KT_REG; ++j) {
        const int kt = KT_LDS + j;
        wreg0[j] = *(const v16bf*)(Wb + ((size_t)(nt0 * 16 + kt) << 9) + lane * 16);
        wreg1[j] = *(const v16bf*)(Wb + ((size_t)(nt1 * 16 + kt) << 9) + lane * 16);
    }

    // ---- init state in registers: x0[b,n] = init_x[n]; emit t = 0 slice ----
    const float bias0 = bias[n0];
    const float bias1 = bias[n1];
    const float iv0 = init_x[n0];
    const float iv1 = init_x[n1];
    v8f x0, x1;
    #pragma unroll
    for (int i = 0; i < 8; ++i) { x0[i] = iv0; x1[i] = iv1; }
    #pragma unroll
    for (int i = 0; i < 8; ++i) {
        const int b = g * 16 + mloc_lo + i;
        out[(size_t)b * NDIM + n0] = x0[i];
        out[(size_t)b * NDIM + n1] = x1[i];
    }

    for (int t = 0; t < T_STEPS; ++t) {
        // Prefetch next step's u rows (32 KB / WG) -> global_prefetch_b8.
        if (t + 1 < T_STEPS) {
            const float* nu = inputs + (size_t)(t + 1) * BATCH * NDIM
                                     + (size_t)g * 16 * NDIM;
            __builtin_prefetch(nu + (size_t)tid * 16, 0, 0);
        }

        // ---- r = tanh(x) from registers -> bf16 row-major in LDS ----
        #pragma unroll
        for (int i = 0; i < 8; ++i) {
            r_lds[mloc_lo + i][n0] = (__bf16)FAST_TANH(x0[i]);
            r_lds[mloc_lo + i][n1] = (__bf16)FAST_TANH(x1[i]);
        }
        __syncthreads();   // all of r visible before any wave's matmul

        // Opaque LDS-W offset: stop LICM from hoisting the streamed slices
        // into registers across the t-loop (round-1 spill failure mode).
        unsigned wofs = (unsigned)(wave * 2) * KT_LDS * 512u;
        asm volatile("" : "+v"(wofs));

        // ---- GEMM: [16 x 512] x [512 x 32] = 2 tiles x 16 K-steps ----
        v8f acc0 = {};
        v8f acc1 = {};
        #pragma unroll
        for (int kt = 0; kt < 16; ++kt) {
            const int k0 = kt * 32 + akoff;
            // A fragment: two contiguous 16 B runs per lane (ds_load_b128 x2).
            v8bf alo = *(const v8bf*)(&r_lds[arow][k0]);
            v8bf ahi = *(const v8bf*)(&r_lds[arow][k0 + 8]);
            v16bf a = __builtin_shufflevector(alo, ahi,
                    0, 1, 2, 3, 4, 5, 6, 7, 8, 9, 10, 11, 12, 13, 14, 15);
            v16bf b0, b1;
            if (kt < KT_LDS) {   // streamed from LDS
                b0 = *(const v16bf*)&w_lds[(wofs + (0 * KT_LDS + kt) * 512u)
                                           + (unsigned)lane * 16u];
                b1 = *(const v16bf*)&w_lds[(wofs + (1 * KT_LDS + kt) * 512u)
                                           + (unsigned)lane * 16u];
            } else {             // register-resident
                b0 = wreg0[kt - KT_LDS];
                b1 = wreg1[kt - KT_LDS];
            }
            acc0 = __builtin_amdgcn_wmma_f32_16x16x32_bf16(
                       false, a, false, b0, (short)0, acc0, false, false);
            acc1 = __builtin_amdgcn_wmma_f32_16x16x32_bf16(
                       false, a, false, b1, (short)0, acc1, false, false);
        }

        // ---- leaky update on register state + write-out ----
        const float* u_row = inputs + (size_t)t * BATCH * NDIM;
        #pragma unroll
        for (int i = 0; i < 8; ++i) {
            const int b = g * 16 + mloc_lo + i;
            {
                float u  = u_row[(size_t)b * NDIM + n0];
                float xn = x0[i] + ALPHA * (-x0[i] + acc0[i] + u + bias0);
                x0[i] = xn;
                out[((size_t)(t + 1) * BATCH + b) * NDIM + n0] = xn;
            }
            {
                float u  = u_row[(size_t)b * NDIM + n1];
                float xn = x1[i] + ALPHA * (-x1[i] + acc1[i] + u + bias1);
                x1[i] = xn;
                out[((size_t)(t + 1) * BATCH + b) * NDIM + n1] = xn;
            }
        }
        __syncthreads();   // everyone done reading r before next overwrite
    }
}

// ---------------------------------------------------------------------------
extern "C" void kernel_launch(void* const* d_in, const int* in_sizes, int n_in,
                              void* d_out, int out_size, void* d_ws, size_t ws_size,
                              hipStream_t stream) {
    const float* inputs         = (const float*)d_in[0];         // [T,B,N]
    const float* W_rec          = (const float*)d_in[1];         // [N,N]
    const unsigned char* mask   = (const unsigned char*)d_in[2]; // [N,N] bool (1 byte)
    const float* bias           = (const float*)d_in[3];         // [1,N]
    const float* init_x         = (const float*)d_in[4];         // [N]
    float* out = (float*)d_out;                                  // [T+1,B,N]

    __bf16* Wb = (__bf16*)d_ws;   // 512 KB packed weights

    pack_w_kernel<<<(NDIM * NDIM + 255) / 256, 256, 0, stream>>>(W_rec, mask, Wb);
    rnn_kernel<<<4, 512, 0, stream>>>(inputs, bias, init_x, Wb, out);
}
